// TSPGNNWithAttention_81853486727224
// MI455X (gfx1250) — matrix operand order repaired and checked by reading, not compiled, not run on hardware
//
#include <hip/hip_runtime.h>
#include <hip/hip_bf16.h>

// ---------------------------------------------------------------------------
// Problem constants (match reference)
// ---------------------------------------------------------------------------
#define NN 10000        // nodes
#define NE 320000       // edges
#define HID 128

typedef __attribute__((ext_vector_type(16))) __bf16 bf16x16;
typedef __attribute__((ext_vector_type(8)))  __bf16 bf16x8;
typedef __attribute__((ext_vector_type(8)))  float  v8f;

// native fp32 -> bf16 (RNE); lowers to v_cvt_pk_bf16_f32
static __device__ __forceinline__ __bf16 f2bf(float f) { return (__bf16)f; }

static __device__ __forceinline__ v8f wmma_bf16(const bf16x16& a, const bf16x16& b, v8f c) {
    return __builtin_amdgcn_wmma_f32_16x16x32_bf16(false, a, false, b, (short)0, c, false, false);
}

static __device__ __forceinline__ float lrelu(float x) { return x >= 0.f ? x : 0.2f * x; }

// order-preserving float<->uint encoding for atomicMax on floats
static __device__ __forceinline__ unsigned fenc(float f) {
    unsigned u = __float_as_uint(f);
    return (u & 0x80000000u) ? ~u : (u | 0x80000000u);
}
static __device__ __forceinline__ float fdec(unsigned u) {
    u = (u & 0x80000000u) ? (u & 0x7FFFFFFFu) : ~u;
    return __uint_as_float(u);
}

// WMMA fragment K-index maps (CDNA5 ISA §7.12.2, 16-bit data)
static __device__ __forceinline__ int kmapA(int i, int hi) { return (i & 7) + ((i >> 3) << 4) + (hi << 3); }
static __device__ __forceinline__ int kmapB(int i, int hi) { return (i & 7) + ((i >> 3) << 4) + (hi << 3); }

static __device__ __forceinline__ v8f zero8() {
    v8f z = {0.f, 0.f, 0.f, 0.f, 0.f, 0.f, 0.f, 0.f};
    return z;
}

// A fragment from global fp32: two contiguous 8-float runs -> 4x b128 loads
static __device__ __forceinline__ void loadA_f32(const float* row, int kb, int hi, bf16x16& a) {
    const float* ap = row + kb + hi * 8;
    float4 q0 = *(const float4*)(ap);
    float4 q1 = *(const float4*)(ap + 4);
    float4 q2 = *(const float4*)(ap + 16);
    float4 q3 = *(const float4*)(ap + 20);
    a[0] = f2bf(q0.x); a[1] = f2bf(q0.y); a[2] = f2bf(q0.z); a[3] = f2bf(q0.w);
    a[4] = f2bf(q1.x); a[5] = f2bf(q1.y); a[6] = f2bf(q1.z); a[7] = f2bf(q1.w);
    a[8]  = f2bf(q2.x); a[9]  = f2bf(q2.y); a[10] = f2bf(q2.z); a[11] = f2bf(q2.w);
    a[12] = f2bf(q3.x); a[13] = f2bf(q3.y); a[14] = f2bf(q3.z); a[15] = f2bf(q3.w);
}

// A fragment from a bf16 LDS row: two 16B runs -> 2x ds_load_b128
static __device__ __forceinline__ bf16x16 loadA_lds(const __bf16* base, int hi) {
    bf16x8 lo = *(const bf16x8*)(base + hi * 8);
    bf16x8 hh = *(const bf16x8*)(base + 16 + hi * 8);
    return __builtin_shufflevector(lo, hh, 0, 1, 2, 3, 4, 5, 6, 7,
                                   8, 9, 10, 11, 12, 13, 14, 15);
}

// ---------------------------------------------------------------------------
// Generic GEMM: C[M,N] = A[M,K] @ B[K,N], fp32 in/out, bf16 WMMA, f32 accum.
// 4 waves / block share one N-tile; B tile staged into LDS with the CDNA5
// async-DMA path (global_load_async_to_lds_b128 + s_wait_asynccnt).
// Requires M % 16 == 0, N % 16 == 0, K % 32 == 0 (true for all uses here).
// ---------------------------------------------------------------------------
__global__ void __launch_bounds__(128)
k_gemm(const float* __restrict__ A, const float* __restrict__ B,
       float* __restrict__ C, int M, int N, int K) {
    __shared__ float sBf[32 * 16];     // fp32 staging tile (2 KB)

    const int tid = threadIdx.x, lane = tid & 31, wv = tid >> 5;
    const int ntiles = N >> 4, mtiles = M >> 4;
    const int tn = blockIdx.x % ntiles;
    const int tg = blockIdx.x / ntiles;
    int tm = tg * 4 + wv;
    if (tm >= mtiles) tm = mtiles - 1;     // tail waves duplicate a tile (benign)
    const int hi = lane >> 4, ml = lane & 15;

    const float* arow = A + (size_t)(tm * 16 + ml) * K;
    const float* bcol = B + tn * 16;

    // per-thread 16B chunk of the 32x16 tile: 128 thr x 4 floats = 512 elems
    const int idx4 = tid * 4;
    const unsigned ldsoff = (unsigned)(size_t)&sBf[idx4];
    const int brow = idx4 >> 4, bcoff = idx4 & 15;

    v8f acc = zero8();
    for (int kb = 0; kb < K; kb += 32) {
        __syncthreads();
        {   // async DMA: one b128 per lane, tracked by ASYNCcnt
            const float* gp = bcol + (size_t)(kb + brow) * N + bcoff;
            asm volatile("global_load_async_to_lds_b128 %0, %1, off"
                         :: "v"(ldsoff), "v"(gp) : "memory");
            asm volatile("s_wait_asynccnt 0x0" ::: "memory");
        }
        __syncthreads();

        bf16x16 a, b;
        loadA_f32(arow, kb, hi, a);
#pragma unroll
        for (int i = 0; i < 16; ++i)
            b[i] = f2bf(sBf[kmapB(i, hi) * 16 + ml]);
        acc = wmma_bf16(a, b, acc);
    }
#pragma unroll
    for (int r = 0; r < 8; ++r)
        C[(size_t)(tm * 16 + r + hi * 8) * N + tn * 16 + ml] = acc[r];
}

// ---------------------------------------------------------------------------
// GAT attention machinery (atomic segment softmax + scatter-add)
// ---------------------------------------------------------------------------
__global__ void k_alpha(const float* __restrict__ h, const float* __restrict__ a_s,
                        const float* __restrict__ a_d, float* __restrict__ as_out,
                        float* __restrict__ ad_out, int H) {
    int t = blockIdx.x * blockDim.x + threadIdx.x;
    if (t >= NN * H) return;
    int n = t / H, hh = t % H;
    const float* hp = h + (size_t)n * H * HID + hh * HID;
    const float* sp = a_s + hh * HID;
    const float* dp = a_d + hh * HID;
    float ss = 0.f, sd = 0.f;
    for (int i = 0; i < HID; i += 4) {
        float4 hv = *(const float4*)(hp + i);
        float4 sv = *(const float4*)(sp + i);
        float4 dv = *(const float4*)(dp + i);
        ss += hv.x * sv.x + hv.y * sv.y + hv.z * sv.z + hv.w * sv.w;
        sd += hv.x * dv.x + hv.y * dv.y + hv.z * dv.z + hv.w * dv.w;
    }
    as_out[t] = ss; ad_out[t] = sd;
}

__global__ void k_init_m(const float* __restrict__ as_, const float* __restrict__ ad_,
                         unsigned* __restrict__ m, int H) {
    int t = blockIdx.x * blockDim.x + threadIdx.x;
    if (t >= NN * H) return;
    m[t] = fenc(lrelu(as_[t] + ad_[t]));          // self loop seeds the max
}

__global__ void k_edge_max(const int* __restrict__ ei, const float* __restrict__ as_,
                           const float* __restrict__ ad_, unsigned* __restrict__ m, int H) {
    int t = blockIdx.x * blockDim.x + threadIdx.x;
    if (t >= NE * H) return;
    int e = t / H, hh = t % H;
    int s = ei[e], d = ei[NE + e];
    float v = lrelu(as_[s * H + hh] + ad_[d * H + hh]);
    atomicMax(&m[d * H + hh], fenc(v));
}

__global__ void k_init_denom(const float* __restrict__ as_, const float* __restrict__ ad_,
                             const unsigned* __restrict__ m, float* __restrict__ den, int H) {
    int t = blockIdx.x * blockDim.x + threadIdx.x;
    if (t >= NN * H) return;
    float es = lrelu(as_[t] + ad_[t]);
    den[t] = __expf(es - fdec(m[t]));             // self-loop contribution
}

__global__ void k_edge_exp(const int* __restrict__ ei, const float* __restrict__ as_,
                           const float* __restrict__ ad_, const unsigned* __restrict__ m,
                           float* __restrict__ den, float* __restrict__ ex, int H) {
    int t = blockIdx.x * blockDim.x + threadIdx.x;
    if (t >= NE * H) return;
    int e = t / H, hh = t % H;
    int s = ei[e], d = ei[NE + e];
    float v = lrelu(as_[s * H + hh] + ad_[d * H + hh]);
    float exv = __expf(v - fdec(m[d * H + hh]));
    ex[t] = exv;
    atomicAdd(&den[d * H + hh], exv);
}

__global__ void k_edge_agg(const int* __restrict__ ei, const float* __restrict__ h,
                           const float* __restrict__ ex, const float* __restrict__ den,
                           float* __restrict__ agg, int H) {
    long idx = (long)blockIdx.x * blockDim.x + threadIdx.x;
    long total = (long)NE * H * HID;
    if (idx >= total) return;
    int e = (int)(idx / (H * HID));
    int c = (int)(idx % (H * HID));
    int hh = c >> 7;
    int s = ei[e], d = ei[NE + e];
    float alpha = ex[e * H + hh] / den[d * H + hh];
    atomicAdd(&agg[(size_t)d * H * HID + c], h[(size_t)s * H * HID + c] * alpha);
}

__global__ void k_finalize(const float* __restrict__ h, const float* __restrict__ agg,
                           const float* __restrict__ as_, const float* __restrict__ ad_,
                           const unsigned* __restrict__ m, const float* __restrict__ den,
                           const float* __restrict__ bias, float* __restrict__ out,
                           int H, int use_relu) {
    long t = (long)blockIdx.x * blockDim.x + threadIdx.x;
    long total = (long)NN * H * HID;
    if (t >= total) return;
    int n = (int)(t / (H * HID));
    int c = (int)(t % (H * HID));
    int hh = c >> 7;
    float es = lrelu(as_[n * H + hh] + ad_[n * H + hh]);
    float aself = __expf(es - fdec(m[n * H + hh])) / den[n * H + hh];
    float v = agg[t] + h[t] * aself + bias[c];
    out[t] = use_relu ? fmaxf(v, 0.f) : (v > 0.f ? v : expm1f(v));
}

// ---------------------------------------------------------------------------
// Fused per-edge pipeline: edge encoder MLP + classifier + log_softmax.
// Block = 128 threads = 4 waves, each wave owns a 16-edge tile (64 edges/blk).
// Weight k-slabs staged once per block into sW and consumed by all 4 waves
// (cuts cls_w1 L2 traffic 4x). K-outer / N-inner keeps 8 accumulators live so
// z1/z2 overwrite the ec tile in place; z2->logits reduced via shfl_xor.
// ---------------------------------------------------------------------------
__global__ void __launch_bounds__(128)
k_fused_edge(const float* __restrict__ EA, const int* __restrict__ EI,
             const float* __restrict__ H3,
             const float* __restrict__ W1, const float* __restrict__ B1,   // enc 16->128
             const float* __restrict__ W2, const float* __restrict__ B2,   // enc 128->128
             const float* __restrict__ CW1, const float* __restrict__ CB1, // cls 384->128
             const float* __restrict__ CW2, const float* __restrict__ CB2, // cls 128->64
             const float* __restrict__ CW3, const float* __restrict__ CB3, // cls 64->2
             float* __restrict__ OUT) {
    __shared__ __bf16 sEC[4][16 * 384];   // per-wave edge-context tile (48 KB)
    __shared__ __bf16 sW[32 * 128];       // shared weight k-slab (8 KB)

    const int tid = threadIdx.x;
    const int w = tid >> 5, lane = tid & 31;
    const int hi = lane >> 4, ml = lane & 15;
    __bf16* ec = sEC[w];
    const int e0 = blockIdx.x * 64 + w * 16;

    // stage rows [kb, kb+32) of a row-major [rows x cols] weight into sW (bf16)
    // 4 consecutive elements per thread: b128 global load + merged bf16 stores
    auto stageW = [&](const float* Wsrc, int rows, int cols, int kb) {
        __syncthreads();
        int tot = 32 * cols;
        for (int idx = tid * 4; idx < tot; idx += 512) {
            int k = kb + (idx / cols);
            int c = idx % cols;
            float4 q;
            if (k < rows) q = *(const float4*)&Wsrc[(size_t)k * cols + c];
            else { q.x = 0.f; q.y = 0.f; q.z = 0.f; q.w = 0.f; }
            sW[idx + 0] = f2bf(q.x); sW[idx + 1] = f2bf(q.y);
            sW[idx + 2] = f2bf(q.z); sW[idx + 3] = f2bf(q.w);
        }
        __syncthreads();
    };

    v8f acc[8];

    // ---- step 1: t1 = relu(ea @ enc_w1 + b1) -> ec cols [0,128) -----------
    {
        stageW(W1, 16, 128, 0);
        bf16x16 a;
        const float* ear = EA + (size_t)(e0 + ml) * 16 + hi * 8;
        float4 q0 = *(const float4*)(ear);
        float4 q1 = *(const float4*)(ear + 4);
        a[0] = f2bf(q0.x); a[1] = f2bf(q0.y); a[2] = f2bf(q0.z); a[3] = f2bf(q0.w);
        a[4] = f2bf(q1.x); a[5] = f2bf(q1.y); a[6] = f2bf(q1.z); a[7] = f2bf(q1.w);
#pragma unroll
        for (int i = 8; i < 16; ++i) a[i] = f2bf(0.f);   // K padded 16->32
#pragma unroll
        for (int nt = 0; nt < 8; ++nt) {
            bf16x16 b;
#pragma unroll
            for (int i = 0; i < 16; ++i) b[i] = sW[kmapB(i, hi) * 128 + nt * 16 + ml];
            acc[nt] = wmma_bf16(a, b, zero8());
        }
#pragma unroll
        for (int nt = 0; nt < 8; ++nt)
#pragma unroll
            for (int r = 0; r < 8; ++r) {
                float v = acc[nt][r] + B1[nt * 16 + ml];
                ec[(r + hi * 8) * 384 + nt * 16 + ml] = f2bf(fmaxf(v, 0.f));
            }
    }

    // ---- step 2: ef = t1 @ enc_w2 + b2 -> ec cols [256,384) ---------------
    {
#pragma unroll
        for (int nt = 0; nt < 8; ++nt) acc[nt] = zero8();
        for (int kb = 0; kb < 128; kb += 32) {
            stageW(W2, 128, 128, kb);
            bf16x16 a = loadA_lds(ec + ml * 384 + kb, hi);
#pragma unroll
            for (int nt = 0; nt < 8; ++nt) {
                bf16x16 b;
#pragma unroll
                for (int i = 0; i < 16; ++i) b[i] = sW[kmapB(i, hi) * 128 + nt * 16 + ml];
                acc[nt] = wmma_bf16(a, b, acc[nt]);
            }
        }
#pragma unroll
        for (int nt = 0; nt < 8; ++nt)
#pragma unroll
            for (int r = 0; r < 8; ++r) {
                float v = acc[nt][r] + B2[nt * 16 + ml];
                ec[(r + hi * 8) * 384 + 256 + nt * 16 + ml] = f2bf(v);
            }
    }

    // ---- step 3: gather h3[src], h3[dst] -> ec cols [0,256) ---------------
    // float4 per iteration: b128 global load + ds_store_b64
    for (int j = lane; j < 16 * 64; j += 32) {
        int m = j >> 6, c4 = j & 63;
        int e = e0 + m;
        int node = (c4 < 32) ? EI[e] : EI[NE + e];
        float4 q = *(const float4*)&H3[(size_t)node * 128 + (c4 & 31) * 4];
        __bf16* dst = ec + m * 384 + c4 * 4;
        dst[0] = f2bf(q.x); dst[1] = f2bf(q.y); dst[2] = f2bf(q.z); dst[3] = f2bf(q.w);
    }

    // ---- step 4: z1 = relu(ec @ cls_w1 + b1) -> ec cols [0,128) -----------
    {
#pragma unroll
        for (int nt = 0; nt < 8; ++nt) acc[nt] = zero8();
        for (int kb = 0; kb < 384; kb += 32) {
            stageW(CW1, 384, 128, kb);
            bf16x16 a = loadA_lds(ec + ml * 384 + kb, hi);
#pragma unroll
            for (int nt = 0; nt < 8; ++nt) {
                bf16x16 b;
#pragma unroll
                for (int i = 0; i < 16; ++i) b[i] = sW[kmapB(i, hi) * 128 + nt * 16 + ml];
                acc[nt] = wmma_bf16(a, b, acc[nt]);
            }
        }
#pragma unroll
        for (int nt = 0; nt < 8; ++nt)
#pragma unroll
            for (int r = 0; r < 8; ++r) {
                float v = acc[nt][r] + CB1[nt * 16 + ml];
                ec[(r + hi * 8) * 384 + nt * 16 + ml] = f2bf(fmaxf(v, 0.f));
            }
    }

    // ---- step 5: z2 = relu(z1 @ cls_w2 + b2), kept in registers -----------
    {
#pragma unroll
        for (int nt = 0; nt < 4; ++nt) acc[nt] = zero8();
        for (int kb = 0; kb < 128; kb += 32) {
            stageW(CW2, 128, 64, kb);
            bf16x16 a = loadA_lds(ec + ml * 384 + kb, hi);
#pragma unroll
            for (int nt = 0; nt < 4; ++nt) {
                bf16x16 b;
#pragma unroll
                for (int i = 0; i < 16; ++i) b[i] = sW[kmapB(i, hi) * 64 + nt * 16 + ml];
                acc[nt] = wmma_bf16(a, b, acc[nt]);
            }
        }
    }

    // ---- step 6: logits (64->2) via shfl reduction + log_softmax ----------
    {
        float p0[8], p1[8];
#pragma unroll
        for (int r = 0; r < 8; ++r) { p0[r] = 0.f; p1[r] = 0.f; }
#pragma unroll
        for (int nt = 0; nt < 4; ++nt) {
            int col = nt * 16 + ml;
            float bias2 = CB2[col];
            float w30 = CW3[col * 2 + 0], w31 = CW3[col * 2 + 1];
#pragma unroll
            for (int r = 0; r < 8; ++r) {
                float z = fmaxf(acc[nt][r] + bias2, 0.f);
                p0[r] += z * w30;
                p1[r] += z * w31;
            }
        }
        // butterfly sum over the 16 lanes of each half (col dimension)
#pragma unroll
        for (int off = 1; off < 16; off <<= 1)
#pragma unroll
            for (int r = 0; r < 8; ++r) {
                p0[r] += __shfl_xor(p0[r], off, 32);
                p1[r] += __shfl_xor(p1[r], off, 32);
            }
        if (ml < 8) {
            int m = ml + hi * 8;                 // row within edge tile
            float l0 = p0[ml] + CB3[0];
            float l1 = p1[ml] + CB3[1];
            float mx = fmaxf(l0, l1);
            float ls = __logf(__expf(l0 - mx) + __expf(l1 - mx));
            OUT[(size_t)(e0 + m) * 2 + 0] = (l0 - mx) - ls;
            OUT[(size_t)(e0 + m) * 2 + 1] = (l1 - mx) - ls;
        }
    }
}

// ---------------------------------------------------------------------------
// Host launcher
// ---------------------------------------------------------------------------
extern "C" void kernel_launch(void* const* d_in, const int* in_sizes, int n_in,
                              void* d_out, int out_size, void* d_ws, size_t ws_size,
                              hipStream_t stream) {
    (void)in_sizes; (void)n_in; (void)out_size; (void)ws_size;

    const float* x    = (const float*)d_in[0];
    const int*   ei   = (const int*)  d_in[1];
    const float* ea   = (const float*)d_in[2];
    const float* g1W  = (const float*)d_in[3];
    const float* g1as = (const float*)d_in[4];
    const float* g1ad = (const float*)d_in[5];
    const float* g1b  = (const float*)d_in[6];
    const float* g2W  = (const float*)d_in[7];
    const float* g2as = (const float*)d_in[8];
    const float* g2ad = (const float*)d_in[9];
    const float* g2b  = (const float*)d_in[10];
    const float* g3W  = (const float*)d_in[11];
    const float* g3as = (const float*)d_in[12];
    const float* g3ad = (const float*)d_in[13];
    const float* g3b  = (const float*)d_in[14];
    const float* encW1 = (const float*)d_in[15];
    const float* encB1 = (const float*)d_in[16];
    const float* encW2 = (const float*)d_in[17];
    const float* encB2 = (const float*)d_in[18];
    // d_in[19..22] = MHA weights: result is discarded by the reference -> skipped
    const float* cW1 = (const float*)d_in[23];
    const float* cB1 = (const float*)d_in[24];
    const float* cW2 = (const float*)d_in[25];
    const float* cB2 = (const float*)d_in[26];
    const float* cW3 = (const float*)d_in[27];
    const float* cB3 = (const float*)d_in[28];

    // workspace layout (floats); total = 23,200,000 floats = 92.8 MB
    float* ws   = (float*)d_ws;
    float* hB   = ws;                    // [10000, 512]  projected features h
    float* aggB = ws + 5120000;          // [10000, 512]  aggregation accumulators
    float* o1   = ws + 10240000;         // [10000, 512]  GAT1 output
    float* o2   = ws + 15360000;         // [10000, 512]  GAT2 output
    float* o3   = ws + 20480000;         // [10000, 128]  GAT3 output
    float* aS   = ws + 21760000;         // [10000, 4]
    float* aD   = aS + 40000;
    float* mB   = aD + 40000;            // encoded-uint segment max
    float* den  = mB + 40000;
    float* exB  = den + 40000;           // [320000, 4]

    auto gemm = [&](const float* A, const float* B, float* C, int M, int N, int K) {
        int mtiles = M / 16;                       // M % 16 == 0 for all uses
        int groups = (mtiles + 3) / 4;
        int blocks = groups * (N / 16);
        k_gemm<<<blocks, 128, 0, stream>>>(A, B, C, M, N, K);
    };

    auto gat = [&](const float* in, int K, int H, const float* W,
                   const float* as_, const float* ad_, const float* b,
                   float* outp, bool relu_act) {
        int NOUT = H * HID;
        gemm(in, W, hB, NN, NOUT, K);
        int nh = NN * H, eh = NE * H;
        k_alpha<<<(nh + 255) / 256, 256, 0, stream>>>(hB, as_, ad_, aS, aD, H);
        k_init_m<<<(nh + 255) / 256, 256, 0, stream>>>(aS, aD, (unsigned*)mB, H);
        k_edge_max<<<(eh + 255) / 256, 256, 0, stream>>>(ei, aS, aD, (unsigned*)mB, H);
        k_init_denom<<<(nh + 255) / 256, 256, 0, stream>>>(aS, aD, (const unsigned*)mB, den, H);
        k_edge_exp<<<(eh + 255) / 256, 256, 0, stream>>>(ei, aS, aD, (const unsigned*)mB, den, exB, H);
        hipMemsetAsync(aggB, 0, (size_t)NN * NOUT * sizeof(float), stream);
        long agg_work = (long)NE * NOUT;
        k_edge_agg<<<(unsigned)((agg_work + 255) / 256), 256, 0, stream>>>(ei, hB, exB, den, aggB, H);
        long fin_work = (long)NN * NOUT;
        k_finalize<<<(unsigned)((fin_work + 255) / 256), 256, 0, stream>>>(
            hB, aggB, aS, aD, (const unsigned*)mB, den, b, outp, H, relu_act ? 1 : 0);
    };

    gat(x,  128, 4, g1W, g1as, g1ad, g1b, o1, false);   // ELU
    gat(o1, 512, 4, g2W, g2as, g2ad, g2b, o2, false);   // ELU
    gat(o2, 512, 1, g3W, g3as, g3ad, g3b, o3, true);    // ReLU

    // fused edge encoder + classifier + log_softmax: 4 waves x 16 edges/block
    k_fused_edge<<<NE / 64, 128, 0, stream>>>(ea, ei, o3,
                                              encW1, encB1, encW2, encB2,
                                              cW1, cB1, cW2, cB2, cW3, cB3,
                                              (float*)d_out);
}
